// Indexer_87462714016260
// MI455X (gfx1250) — matrix operand order repaired and checked by reading, not compiled, not run on hardware
//
#include <hip/hip_runtime.h>
#include <stdint.h>

// ---------------------------------------------------------------------------
// Problem constants (from reference)
// ---------------------------------------------------------------------------
#define BSZ    2
#define SEQ    1024
#define DIM    2048
#define NKV    2048
#define NH     32
#define HD     128
#define NQ     (NH * HD)      // 4096
#define TOPK_N 1024
#define RSQRT_HD 0.08838834764831845f   // 128^-0.5
#define RSQRT_NH 0.17677669529663687f   // 32^-0.5

// ---------------------------------------------------------------------------
// CDNA5 WMMA types
// ---------------------------------------------------------------------------
typedef __bf16 bf16_t;
typedef bf16_t v16bf __attribute__((ext_vector_type(16)));
typedef float  v8f   __attribute__((ext_vector_type(8)));
typedef unsigned int v4u __attribute__((ext_vector_type(4)));
typedef unsigned int v8u __attribute__((ext_vector_type(8)));

struct __align__(16) U4 { unsigned int a, b, c, d; };
union FragCast { U4 u[2]; v16bf v; };

// A-fragment (16x32 bf16, M x K) from a row-major [row][k] array.
// Per ISA 7.12.2: lanes 0-15 hold M=lane, K chunks {0..7, 16..23};
// lanes 16-31 hold M=lane-16, K chunks {8..15, 24..31}.
__device__ __forceinline__ v16bf load_frag_a(const bf16_t* base, int rowStride,
                                             int lane, int kstep) {
  const int row = lane & 15;
  const int c0  = kstep + ((lane & 16) ? 8 : 0);
  const bf16_t* p = base + (size_t)row * rowStride + c0;
  FragCast f;
  f.u[0] = *(const U4*)(p);        // K chunk +0..7
  f.u[1] = *(const U4*)(p + 16);   // K chunk +16..23
  return f.v;
}

// B-fragment (32x16 bf16, K x N) from a row-major [n][k] array (i.e. B^T rows).
// lanes 0-15: N=lane, K chunks {0..7, 16..23}; lanes 16-31: K chunks {8..15, 24..31}.
__device__ __forceinline__ v16bf load_frag_b(const bf16_t* base, int rowStride,
                                             int lane, int kstep) {
  const int col = lane & 15;
  const int c0  = kstep + ((lane & 16) ? 8 : 0);
  const bf16_t* p = base + (size_t)col * rowStride + c0;
  FragCast f;
  f.u[0] = *(const U4*)(p);
  f.u[1] = *(const U4*)(p + 16);
  return f.v;
}

#define WMMA_BF16(a, b, c) \
  __builtin_amdgcn_wmma_f32_16x16x32_bf16(false, (a), false, (b), (short)0, (c), false, false)

// ---------------------------------------------------------------------------
// Tensor Data Mover: async 2D tile load global->LDS (CDNA5 TDM, ISA ch.8).
// rows x rowElems tile of 2-byte elements, global row pitch strideElems,
// optional LDS padding of 16B after every 256B row (pad_interval=5 -> 64
// dwords, pad_amount=3 -> 4 dwords) for bank-conflict-free ds_load_b128.
// ---------------------------------------------------------------------------
__device__ __forceinline__ void tdm_load_2d(unsigned ldsOff, const void* gptr,
                                            unsigned rows, unsigned rowElems,
                                            unsigned strideElems, bool padRows) {
  uint64_t ga = (uint64_t)(uintptr_t)gptr;
  v4u g0 = {1u,                                            // count=1
            ldsOff,                                        // lds_addr
            (unsigned)(ga & 0xFFFFFFFFu),                  // global_addr[31:0]
            (unsigned)((ga >> 32) & 0x1FFFFFFu) | (2u << 30)}; // addr[56:32]|type=2
  unsigned w0 = 0x00010000u;                               // data_size=1 (2B)
  if (padRows) w0 |= (1u << 20) | (5u << 22) | (3u << 25); // pad 4 dwords / 64 dwords
  v8u g1 = {w0,
            (rowElems & 0xFFFFu) << 16,                    // tensor_dim0[15:0] @ 63:48
            ((rowElems >> 16) & 0xFFFFu) | ((rows & 0xFFFFu) << 16), // dim0 hi | dim1 lo
            ((rows >> 16) & 0xFFFFu) | ((rowElems & 0xFFFFu) << 16), // dim1 hi | tile_dim0
            rows & 0xFFFFu,                                // tile_dim1
            strideElems,                                   // tensor_dim0_stride[31:0]
            0u, 0u};
  asm volatile("tensor_load_to_lds %0, %1" :: "s"(g0), "s"(g1) : "memory");
}

// ---------------------------------------------------------------------------
// Kernel A: fold per-head Hadamard (Sylvester: H[i][j] = (-1)^popc(i&j)) and
// 128^-0.5 into wq; store transposed bf16: wqrotT[n][k], n = h*128+j, k = dim.
// ---------------------------------------------------------------------------
__global__ void rotwq_kernel(const float* __restrict__ wq, bf16_t* __restrict__ wqrotT) {
  const int k = blockIdx.x;   // 0..2047
  const int h = blockIdx.y;   // 0..31
  const int j = threadIdx.x;  // 0..127
  __shared__ float row[HD];
  row[j] = wq[(size_t)k * NQ + h * HD + j];
  __syncthreads();
  float s = 0.f;
  #pragma unroll 8
  for (int i = 0; i < HD; ++i) {
    float v = row[i];
    s += (__popc(i & j) & 1) ? -v : v;
  }
  s *= RSQRT_HD;
  wqrotT[(size_t)(h * HD + j) * DIM + k] = (bf16_t)s;
}

// ---------------------------------------------------------------------------
// Kernel B: q = x @ wqrotT^T via bf16 WMMA, fused RoPE epilogue.
// grid (16, 64, 2), block 256 (8 waves). Each wave: two 16x16 output tiles.
// LDS: 16 x 1024 bf16 x-tile (restaged for each K half), padded rows.
// ---------------------------------------------------------------------------
#define XT_STRIDE 1032   // 1024 + 8 pad; row pitch 2064B = 129*16B (16B aligned)

__device__ __forceinline__ void rope_store_q(v8f acc, int b, int s0, int n0, int lane,
                                             const float* __restrict__ cosT,
                                             const float* __restrict__ sinT,
                                             bf16_t* __restrict__ qOut) {
  const int N = lane & 15;
  const int n = n0 + N;
  const int h = n >> 7;
  const int d = n & 127;
  const int j = d >> 1;
  const int mOff = (lane & 16) ? 8 : 0;
  #pragma unroll
  for (int r = 0; r < 8; ++r) {
    const int s = s0 + r + mOff;
    float v = acc[r];
    float p = __shfl_xor(v, 1, 32);   // partner element of the RoPE pair
    float cs = cosT[(size_t)s * (HD / 2) + j];
    float sn = sinT[(size_t)s * (HD / 2) + j];
    float out = (d & 1) ? (p * sn + v * cs) : (v * cs - p * sn);
    qOut[(((size_t)(b * SEQ) + s) * NH + h) * HD + d] = (bf16_t)out;
  }
}

__global__ void qgemm_kernel(const float* __restrict__ x, const bf16_t* __restrict__ wqrotT,
                             const float* __restrict__ cosT, const float* __restrict__ sinT,
                             bf16_t* __restrict__ qOut) {
  const int b = blockIdx.z, sT = blockIdx.y, nB = blockIdx.x;
  const int s0 = sT * 16, nBase = nB * 256;
  const int tid = threadIdx.x, wave = tid >> 5, lane = tid & 31;
  __shared__ __align__(16) bf16_t xT[16 * XT_STRIDE];

  const int n0a = nBase + wave * 32;
  const int n0b = n0a + 16;
  v8f acc0 = {0, 0, 0, 0, 0, 0, 0, 0};
  v8f acc1 = {0, 0, 0, 0, 0, 0, 0, 0};

  for (int half = 0; half < 2; ++half) {
    // stage 16 x 1024 fp32 -> bf16
    for (int e = tid; e < 16 * 1024; e += 256) {
      const int r = e >> 10, c = e & 1023;
      xT[r * XT_STRIDE + c] =
          (bf16_t)x[((size_t)(b * SEQ) + s0 + r) * DIM + half * 1024 + c];
    }
    __syncthreads();
    const bf16_t* bBase0 = wqrotT + (size_t)n0a * DIM + half * 1024;
    const bf16_t* bBase1 = wqrotT + (size_t)n0b * DIM + half * 1024;
    #pragma unroll 4
    for (int k = 0; k < 1024; k += 32) {
      v16bf af = load_frag_a(xT, XT_STRIDE, lane, k);
      v16bf b0 = load_frag_b(bBase0, DIM, lane, k);
      v16bf b1 = load_frag_b(bBase1, DIM, lane, k);
      acc0 = WMMA_BF16(af, b0, acc0);
      acc1 = WMMA_BF16(af, b1, acc1);
    }
    __syncthreads();
  }
  rope_store_q(acc0, b, s0, n0a, lane, cosT, sinT, qOut);
  rope_store_q(acc1, b, s0, n0b, lane, cosT, sinT, qOut);
}

// ---------------------------------------------------------------------------
// Kernel C: k preparation. Gather from paged cache, FP8-e4m3 quant-dequant
// (software RNE), in-LDS fast Walsh-Hadamard transform, RoPE by pos. -> bf16.
// One block of 128 threads per (n, b); thread = element d.
// ---------------------------------------------------------------------------
__global__ void kprep_kernel(const float* __restrict__ cache,
                             const int* __restrict__ block_offsets,
                             const int* __restrict__ positions,
                             const float* __restrict__ cosT, const float* __restrict__ sinT,
                             bf16_t* __restrict__ kOut) {
  const int n = blockIdx.x, b = blockIdx.y, d = threadIdx.x;
  __shared__ float v[HD];
  __shared__ float red[HD];

  const int pos  = positions[b * NKV + n];
  const int safe = pos < 0 ? 0 : pos;
  const int bi   = safe >> 4;               // safe*4/64
  const bool valid = (pos >= 0) && (bi < 128);
  const int sbi  = bi < 127 ? bi : 127;
  const int off  = safe & 15;               // safe % cache.shape[1](=16)
  const int phys = block_offsets[b * 128 + sbi];
  float val = valid ? cache[((size_t)phys * 16 + off) * HD + d] : 0.f;

  // per-row max-abs -> fp8 scale
  red[d] = fabsf(val);
  __syncthreads();
  #pragma unroll
  for (int st = 64; st > 0; st >>= 1) {
    if (d < st) red[d] = fmaxf(red[d], red[d + st]);
    __syncthreads();
  }
  const float scale = fmaxf(red[0], 1e-4f) * (1.f / 448.f);

  // software e4m3 quant-dequant, round-to-nearest-even
  {
    float y = val / scale;
    float a = fabsf(y);
    float qa;
    if (a >= 0.015625f) {                       // normal range (>= 2^-6)
      unsigned ub = __float_as_uint(a);
      unsigned r  = ub + 0x7FFFFu + ((ub >> 20) & 1u);  // RNE at 3 mantissa bits
      qa = __uint_as_float(r & 0xFFF00000u);
      if (qa > 448.f) qa = 448.f;
    } else {                                    // subnormals: multiples of 2^-9
      qa = rintf(a * 512.f) * (1.f / 512.f);
    }
    val = copysignf(qa, y) * scale;
  }

  // FWHT over 128 elements == row @ Hadamard(128)
  v[d] = val;
  __syncthreads();
  #pragma unroll
  for (int st = 1; st < HD; st <<= 1) {
    float p = v[d ^ st];
    float m = v[d];
    float nv = (d & st) ? (p - m) : (m + p);
    __syncthreads();
    v[d] = nv;
    __syncthreads();
  }
  val = v[d] * RSQRT_HD;

  // RoPE at pos
  v[d] = val;
  __syncthreads();
  {
    float p = v[d ^ 1];
    const int j = d >> 1;
    float cs = cosT[(size_t)safe * (HD / 2) + j];
    float sn = sinT[(size_t)safe * (HD / 2) + j];
    val = (d & 1) ? (p * sn + val * cs) : (val * cs - p * sn);
  }
  kOut[((size_t)(b * NKV) + n) * HD + d] = (bf16_t)val;
}

// ---------------------------------------------------------------------------
// Kernel D: gate w = (x @ wgate) * 32^-0.5.  Block 32 per (s,b), lane = head.
// x element broadcast across lanes, wgate reads coalesced.
// ---------------------------------------------------------------------------
__global__ void gate_kernel(const float* __restrict__ x, const float* __restrict__ wgate,
                            float* __restrict__ wOut) {
  const int s = blockIdx.x, b = blockIdx.y, h = threadIdx.x;
  const float* xr = x + ((size_t)(b * SEQ) + s) * DIM;
  float acc = 0.f;
  #pragma unroll 8
  for (int k = 0; k < DIM; ++k) acc += xr[k] * wgate[(size_t)k * NH + h];
  wOut[((size_t)(b * SEQ) + s) * NH + h] = acc * RSQRT_NH;
}

// ---------------------------------------------------------------------------
// Kernel E: fused idx_score = sum_h relu((q_h . k^T) * 128^-0.5) * w_h.
// grid (64 sTiles, 2), block 256 (8 waves). The q s-tile (16 x 4096 bf16,
// 131 KB incl. bank-skew padding) is staged in LDS ONCE; the sixteen
// 128x128 bf16 k-chunks are streamed through a double-buffered TDM pipeline
// (tensor_load_to_lds chunk c+1 overlapped with WMMA on chunk c,
// s_wait_tensorcnt 1). TDM row padding gives the k buffers a 272B pitch so
// both fragment loads are bank-conflict free. 203 KB LDS total (<320 KB/WGP).
// scores tensor (536 MB) is never materialized; q global traffic drops
// 16x vs. the per-chunk-grid version (256 MB -> 16 MB).
// ---------------------------------------------------------------------------
#define QT_STRIDE 4104                    // 4096 + 8 -> 8208B pitch = 513*16B
#define KB_STRIDE 136                     // 128 + 8  -> 272B pitch (TDM pad)
#define KBUF_ELEMS (128 * KB_STRIDE)      // 17408 elems = 34816 B per buffer

struct __align__(16) ScoresSmem {
  bf16_t kBuf[2 * KBUF_ELEMS];            // at LDS offset 0 (first member)
  bf16_t qT[16 * QT_STRIDE];
  float  wT[16 * NH];
};

__global__ void scores_kernel(const bf16_t* __restrict__ qB, const bf16_t* __restrict__ kB,
                              const float* __restrict__ wG, float* __restrict__ iscore) {
  const int b = blockIdx.y, sT = blockIdx.x;
  const int s0 = sT * 16;
  const int tid = threadIdx.x, wave = tid >> 5, lane = tid & 31;
  __shared__ ScoresSmem sm;

  // stage gate weights (16 x 32 f32)
  for (int e = tid; e < 16 * NH; e += 256)
    sm.wT[e] = wG[((size_t)(b * SEQ) + s0 + (e >> 5)) * NH + (e & 31)];

  // stage q s-tile: 16 rows x 4096 bf16, vectorized 8-elem copies
  for (int e = tid; e < 16 * (NQ / 8); e += 256) {
    const int r = e >> 9, cc = e & 511;   // 512 8-elem chunks per row
    *(U4*)(&sm.qT[r * QT_STRIDE + cc * 8]) =
        *(const U4*)(qB + ((size_t)(b * SEQ) + s0 + r) * NQ + cc * 8);
  }

  const bf16_t* kBase = kB + (size_t)(b * NKV) * HD;
  // prologue: TDM chunk 0 -> buffer 0 (kBuf sits at LDS offset 0)
  if (wave == 0) tdm_load_2d(0u, kBase, 128, HD, HD, true);

  const int N = lane & 15;
  const int mOff = (lane & 16) ? 8 : 0;

  for (int c = 0; c < 16; ++c) {
    if (wave == 0) {
      if (c + 1 < 16) {
        // prefetch next chunk into the other buffer, keep one in flight
        const unsigned dstOff = (c & 1) ? 0u : (unsigned)(KBUF_ELEMS * sizeof(bf16_t));
        tdm_load_2d(dstOff, kBase + (size_t)(c + 1) * 128 * HD, 128, HD, HD, true);
        __builtin_amdgcn_s_wait_tensorcnt(1);   // chunk c complete (in-order)
      } else {
        __builtin_amdgcn_s_wait_tensorcnt(0);
      }
    }
    __syncthreads();   // chunk c visible to all waves (qT/wT staged too at c==0)

    const bf16_t* kTile = sm.kBuf + (c & 1) * KBUF_ELEMS + (wave * 16) * KB_STRIDE;
    v8f acc = {0, 0, 0, 0, 0, 0, 0, 0};
    for (int h = 0; h < NH; ++h) {
      v8f dacc = {0, 0, 0, 0, 0, 0, 0, 0};
      const bf16_t* qBase = sm.qT + h * HD;
      #pragma unroll
      for (int ks = 0; ks < HD; ks += 32) {
        v16bf af  = load_frag_a(qBase, QT_STRIDE, lane, ks);
        v16bf bfg = load_frag_b(kTile, KB_STRIDE, lane, ks);
        dacc = WMMA_BF16(af, bfg, dacc);
      }
      #pragma unroll
      for (int r = 0; r < 8; ++r) {
        float sc = dacc[r] * RSQRT_HD;
        sc = sc > 0.f ? sc : 0.f;                     // relu
        acc[r] += sc * sm.wT[(r + mOff) * NH + h];    // gate weight, row s0+M
      }
    }
    const int n0 = c * 128 + wave * 16;
    #pragma unroll
    for (int r = 0; r < 8; ++r)
      iscore[((size_t)(b * SEQ) + s0 + r + mOff) * NKV + n0 + N] = acc[r];
    __syncthreads();   // all waves done with buffer (c&1) before TDM reuses it
  }
}

// ---------------------------------------------------------------------------
// Kernel F: exact sorted top-1024 of 2048 per (b,s): full bitonic sort in LDS
// on (val desc, idx asc) keys -- matches lax.top_k ordering/tie-break.
// Block 1024 threads (32 waves), 16 KB LDS.
// ---------------------------------------------------------------------------
__global__ void topk_kernel(const float* __restrict__ iscore,
                            float* __restrict__ outVals, int* __restrict__ outIdx) {
  const int s = blockIdx.x, b = blockIdx.y, t = threadIdx.x;
  __shared__ float sv[NKV];
  __shared__ int   si[NKV];
  const float* row = iscore + ((size_t)(b * SEQ) + s) * NKV;
  sv[t] = row[t];               si[t] = t;
  sv[t + 1024] = row[t + 1024]; si[t + 1024] = t + 1024;
  __syncthreads();

  for (int k = 2; k <= NKV; k <<= 1) {
    for (int j = k >> 1; j > 0; j >>= 1) {
      const int i = ((t & ~(j - 1)) << 1) | (t & (j - 1));
      const int l = i | j;
      const bool up = ((i & k) == 0);
      float av = sv[i], bv = sv[l];
      int   ai = si[i], bi = si[l];
      // "a precedes b" in output order: larger value first, then lower index
      const bool lFirst = (bv > av) || (bv == av && bi < ai);
      const bool iFirst = (av > bv) || (av == bv && ai < bi);
      if (up ? lFirst : iFirst) {
        sv[i] = bv; sv[l] = av;
        si[i] = bi; si[l] = ai;
      }
      __syncthreads();
    }
  }
  const size_t o = ((size_t)(b * SEQ) + s) * TOPK_N + t;
  outVals[o] = sv[t];
  outIdx[o]  = si[t];
}

// ---------------------------------------------------------------------------
// Host launcher
// ---------------------------------------------------------------------------
extern "C" void kernel_launch(void* const* d_in, const int* in_sizes, int n_in,
                              void* d_out, int out_size, void* d_ws, size_t ws_size,
                              hipStream_t stream) {
  (void)in_sizes; (void)n_in; (void)out_size; (void)ws_size;
  const float* x         = (const float*)d_in[0];
  const float* wq        = (const float*)d_in[1];
  const float* wgate     = (const float*)d_in[2];
  const float* cache     = (const float*)d_in[3];
  const float* cosT      = (const float*)d_in[4];
  const float* sinT      = (const float*)d_in[5];
  const int*   blockOff  = (const int*)d_in[6];
  const int*   positions = (const int*)d_in[7];

  // workspace layout (bytes): ~50 MiB total
  char* ws = (char*)d_ws;
  bf16_t* wqrotT = (bf16_t*)(ws);                              // 16 MiB: [4096][2048] bf16
  bf16_t* qB     = (bf16_t*)(ws + (16u << 20));                // 16 MiB: [2][1024][32][128] bf16
  bf16_t* kB     = (bf16_t*)(ws + (32u << 20));                // 1 MiB : [2][2048][128] bf16
  float*  wOut   = (float*) (ws + (33u << 20));                // 256 KiB: [2][1024][32] f32
  float*  iscore = (float*) (ws + (34u << 20));                // 16 MiB: [2][1024][2048] f32

  float* outVals = (float*)d_out;                              // (2,1024,1024) f32
  int*   outIdx  = (int*)((float*)d_out + (size_t)BSZ * SEQ * TOPK_N);

  rotwq_kernel <<<dim3(DIM, NH),        128,  0, stream>>>(wq, wqrotT);
  qgemm_kernel <<<dim3(NQ / 256, SEQ / 16, BSZ), 256, 0, stream>>>(x, wqrotT, cosT, sinT, qB);
  kprep_kernel <<<dim3(NKV, BSZ),       HD,   0, stream>>>(cache, blockOff, positions, cosT, sinT, kB);
  gate_kernel  <<<dim3(SEQ, BSZ),       NH,   0, stream>>>(x, wgate, wOut);
  scores_kernel<<<dim3(SEQ / 16, BSZ),  256,  0, stream>>>(qB, kB, wOut, iscore);
  topk_kernel  <<<dim3(SEQ, BSZ),       1024, 0, stream>>>(iscore, outVals, outIdx);
}
